// ChebyshevKANLayer_30666066494086
// MI455X (gfx1250) — matrix operand order repaired and checked by reading, not compiled, not run on hardware
//
#include <hip/hip_runtime.h>
#include <cmath>

typedef __attribute__((ext_vector_type(16))) _Float16 v16h;
typedef __attribute__((ext_vector_type(8)))  float    v8f;
typedef __attribute__((ext_vector_type(4)))  float    v4f;
typedef __attribute__((ext_vector_type(4)))  int      v4i;

#define INPUT_DIM  512
#define OUT_DIM    32
#define NDEG       8          // degrees 1..8 via WMMA; degree 0 folded into bias
#define NCHUNK     16         // 512 features / 32 per K-chunk
#define MT         2          // M-tiles (16 rows each) per wave
#define WAVES      8
#define ROWS_PER_BLOCK (WAVES * MT * 16)            // 256
#define CHUNK_F16  (NDEG * 2 * 32 * 16)             // 8192 f16 = 16 KB per chunk
#define CHUNK_V4   (CHUNK_F16 * 2 / 16)             // 1024 v4f per chunk
#define WSW_F16    (NCHUNK * CHUNK_F16)             // 131072 f16 = 256 KB
#define WSW_BYTES  (WSW_F16 * 2)

#if __has_builtin(__builtin_amdgcn_global_load_async_to_lds_b128) && \
    __has_builtin(__builtin_amdgcn_s_wait_asynccnt)
#define ASYNC_STAGE 1
#endif

// ---------------------------------------------------------------------------
// fast tanh: prefer gfx1250 hardware v_tanh_f32; fallback exp2-based.
// ---------------------------------------------------------------------------
__device__ __forceinline__ float fast_tanh(float x) {
#if __has_builtin(__builtin_amdgcn_tanhf)
  return __builtin_amdgcn_tanhf(x);
#elif __has_builtin(__builtin_amdgcn_tanh_f32)
  return __builtin_amdgcn_tanh_f32(x);
#else
  // tanh(x) = 1 - 2 / (exp(2x) + 1); exp(2x) = exp2(2*log2(e)*x)
  float t = x * 2.8853900817779268f;
#if __has_builtin(__builtin_amdgcn_exp2f)
  float e = __builtin_amdgcn_exp2f(t);
#else
  float e = exp2f(t);
#endif
#if __has_builtin(__builtin_amdgcn_rcpf)
  float r = __builtin_amdgcn_rcpf(e + 1.0f);
#else
  float r = 1.0f / (e + 1.0f);
#endif
  return __builtin_fmaf(-2.0f, r, 1.0f);
#endif
}

// ---------------------------------------------------------------------------
// Stage one 16 KB W chunk global->LDS. Async path uses the CDNA5 async-copy
// engine (no VGPR payload, tracked by ASYNCcnt); fallback is load+ds_store.
// Builtin signature (from hipcc diagnostic): param0 = int4 addrspace(1)*,
// so cast through v4i with const dropped.
// ---------------------------------------------------------------------------
__device__ __forceinline__ void stage_chunk(const v4f* __restrict__ src,
                                            v4f* dst, int tid) {
#ifdef ASYNC_STAGE
  typedef __attribute__((address_space(1))) v4i* gv4i;
  typedef __attribute__((address_space(3))) v4i* lv4i;
#pragma unroll
  for (int s = 0; s < 4; ++s)
    __builtin_amdgcn_global_load_async_to_lds_b128(
        (gv4i)(v4i*)(src + tid + 256 * s),
        (lv4i)(v4i*)(dst + tid + 256 * s), 0, 0);
#else
#pragma unroll
  for (int s = 0; s < 4; ++s) dst[tid + 256 * s] = src[tid + 256 * s];
#endif
}

__device__ __forceinline__ void stage_wait() {
#ifdef ASYNC_STAGE
  __builtin_amdgcn_s_wait_asynccnt(0);
#endif
}

// ---------------------------------------------------------------------------
// Prepass: swizzle cheby_coeffs (fp32 [512][32][9]) into f16 B-fragment blobs
// and compute the degree-0 column-sum bias.
// Blob flat index: ((((c*8 + j)*2 + t)*32 + lane)*16 + e), f16 elements.
// B-matrix (32x16 f16) lane layout: lane l holds column n = l&15;
//   lanes 0-15 hold K = {0..7,16..23}, lanes 16-31 hold K = {8..15,24..31}.
// ---------------------------------------------------------------------------
__global__ __launch_bounds__(256) void cheby_kan_prep(
    const float* __restrict__ w, _Float16* __restrict__ wsw,
    float* __restrict__ bias)
{
  const int idx = blockIdx.x * 256 + threadIdx.x;   // 0 .. WSW_F16-1
  const int e = idx & 15;
  const int l = (idx >> 4) & 31;
  const int t = (idx >> 9) & 1;
  const int j = (idx >> 10) & 7;
  const int c = idx >> 13;
  const int n = t * 16 + (l & 15);
  const int k = (e < 8) ? (l >> 4) * 8 + e : 16 + (l >> 4) * 8 + (e - 8); // K within chunk
  const int i = c * 32 + k;                         // input feature
  wsw[idx] = (_Float16)w[((size_t)i * OUT_DIM + n) * 9 + (j + 1)];

  if (idx < OUT_DIM) {                              // degree-0 bias
    float s = 0.0f;
    for (int i2 = 0; i2 < INPUT_DIM; ++i2)
      s += w[((size_t)i2 * OUT_DIM + idx) * 9];
    bias[idx] = s;
  }
}

// ---------------------------------------------------------------------------
// Main kernel: on-the-fly Chebyshev features -> v_wmma_f32_16x16x32_f16.
// 8 waves/block; each wave computes a 32x32 output tile (2 M-tiles x 2 N-tiles).
// W chunks double-buffered through LDS via async copies; x read once (NT).
// ---------------------------------------------------------------------------
__global__ __launch_bounds__(256) void cheby_kan_wmma(
    const float* __restrict__ x, const _Float16* __restrict__ wsw,
    const float* __restrict__ bias, float* __restrict__ y)
{
  __shared__ v16h lds_b[2][NDEG][2][32];            // 32 KB, double buffered

  const int tid  = threadIdx.x;
  const int wave = tid >> 5;
  const int lane = tid & 31;
  const int h    = lane >> 4;                       // lane half
  const int ln   = lane & 15;

  const int waveRow = blockIdx.x * ROWS_PER_BLOCK + wave * (MT * 16);

  v8f acc[MT][2];
#pragma unroll
  for (int mt = 0; mt < MT; ++mt)
#pragma unroll
    for (int t = 0; t < 2; ++t)
#pragma unroll
      for (int e = 0; e < 8; ++e) acc[mt][t][e] = 0.0f;

  // stage chunk 0 into buffer 0
  stage_chunk((const v4f*)wsw, (v4f*)&lds_b[0][0][0][0], tid);
  stage_wait();
  __syncthreads();

#pragma unroll 1
  for (int c = 0; c < NCHUNK; ++c) {
    const int cb = c & 1;
    if (c + 1 < NCHUNK) {                           // prefetch next W chunk
      stage_chunk((const v4f*)wsw + (size_t)(c + 1) * CHUNK_V4,
                  (v4f*)&lds_b[cb ^ 1][0][0][0], tid);
    }

    // ---- load x in A-fragment layout, compute tanh, init recurrence ----
    // A-matrix (16x32 f16): lane<16 row=ln holds K {0..7,16..23};
    //                       lane>=16 row=ln holds K {8..15,24..31}.
    float xt[MT][16], prev[MT][16], cur[MT][16];
#pragma unroll
    for (int mt = 0; mt < MT; ++mt) {
      const int row = waveRow + mt * 16 + ln;
      const v4f* xr = (const v4f*)(x + (size_t)row * INPUT_DIM + c * 32 + h * 8);
      if (c + 1 < NCHUNK)                           // pull next x block toward L2
        __builtin_prefetch((const char*)(xr) + 32 * sizeof(float), 0, 0);
      v4f q0 = __builtin_nontemporal_load(xr + 0);
      v4f q1 = __builtin_nontemporal_load(xr + 1);
      v4f q2 = __builtin_nontemporal_load(xr + 4);  // +16 floats
      v4f q3 = __builtin_nontemporal_load(xr + 5);
      float xv[16] = {q0.x, q0.y, q0.z, q0.w, q1.x, q1.y, q1.z, q1.w,
                      q2.x, q2.y, q2.z, q2.w, q3.x, q3.y, q3.z, q3.w};
#pragma unroll
      for (int e = 0; e < 16; ++e) {
        float tv = fast_tanh(xv[e]);
        xt[mt][e]   = tv;
        prev[mt][e] = 1.0f;        // U0
        cur[mt][e]  = 2.0f * tv;   // U1
      }
    }

    // ---- degrees 1..8: pack A, WMMA against both N-tiles, recur ----
#pragma unroll
    for (int j = 0; j < NDEG; ++j) {
      const v16h b0 = lds_b[cb][j][0][lane];
      const v16h b1 = lds_b[cb][j][1][lane];
#pragma unroll
      for (int mt = 0; mt < MT; ++mt) {
        v16h a;
#pragma unroll
        for (int e = 0; e < 16; ++e) a[e] = (_Float16)cur[mt][e];
        acc[mt][0] = __builtin_amdgcn_wmma_f32_16x16x32_f16(
            false, a, false, b0, (short)0, acc[mt][0], false, false);
        acc[mt][1] = __builtin_amdgcn_wmma_f32_16x16x32_f16(
            false, a, false, b1, (short)0, acc[mt][1], false, false);
        if (j + 1 < NDEG) {
#pragma unroll
          for (int e = 0; e < 16; ++e) {
            float nx = __builtin_fmaf(2.0f * xt[mt][e], cur[mt][e], -prev[mt][e]);
            prev[mt][e] = cur[mt][e];
            cur[mt][e]  = nx;
          }
        }
      }
    }
    stage_wait();      // async staging of chunk c+1 landed in LDS
    __syncthreads();   // all waves done with buffer cb; staged data visible
  }

  // ---- add degree-0 bias, store ----
  // C/D 16x16 f32 layout: lane<16: N=ln, VGPR e -> M=e; lane>=16: M=8+e.
#pragma unroll
  for (int t = 0; t < 2; ++t) {
    const int n = t * 16 + ln;
    const float bb = bias[n];
#pragma unroll
    for (int mt = 0; mt < MT; ++mt) {
      const int rbase = waveRow + mt * 16 + h * 8;
#pragma unroll
      for (int e = 0; e < 8; ++e)
        __builtin_nontemporal_store(acc[mt][t][e] + bb,
                                    &y[(size_t)(rbase + e) * OUT_DIM + n]);
    }
  }
}

// ---------------------------------------------------------------------------
extern "C" void kernel_launch(void* const* d_in, const int* in_sizes, int n_in,
                              void* d_out, int out_size, void* d_ws, size_t ws_size,
                              hipStream_t stream) {
  const float* x = (const float*)d_in[0];          // [65536, 512] f32
  const float* w = (const float*)d_in[1];          // [512, 32, 9] f32
  float* y = (float*)d_out;                        // [65536, 32] f32

  _Float16* wsw = (_Float16*)d_ws;                 // 256 KB swizzled f16 W
  float* bias = (float*)((char*)d_ws + WSW_BYTES); // 32 f32 degree-0 bias

  cheby_kan_prep<<<WSW_F16 / 256, 256, 0, stream>>>(w, wsw, bias);

  const int nrows = in_sizes[0] / INPUT_DIM;       // 65536
  cheby_kan_wmma<<<nrows / ROWS_PER_BLOCK, 256, 0, stream>>>(x, wsw, bias, y);
}
